// GAT_8504035246554
// MI455X (gfx1250) — compile-verified, matrix operand order checked by reading, not run on hardware
//
#include <hip/hip_runtime.h>
#include <hip/hip_bf16.h>

// ---------------------------------------------------------------------------
// GAT forward (2 layers, A and A^T branches) for MI455X / gfx1250.
//
// Strategy (see analysis): compress dense 0/1 adjacency to bitmasks once
// (256MB -> 2x8MB), precompute exact per-row softmax max analytically from
// the rank-1 logit structure, then run the dominant attn@Wh products as
// streaming bf16 WMMA (v_wmma_f32_16x16x32_bf16) with row sums obtained by
// a ones-column appended to the transposed Wh panel.
// ---------------------------------------------------------------------------

typedef __attribute__((ext_vector_type(16))) __bf16 v16bf;
typedef __attribute__((ext_vector_type(8)))  float  v8f;

#define NODES 8192
#define MASK_WORDS 256          // 8192 / 32
#define NEG_BIG (-9.0e15f)

__device__ __forceinline__ float leaky01(float t) { return t > 0.f ? t : 0.1f * t; }

// ---------------------------------------------------------------------------
// 1) Adjacency -> bitmask (one wave per 32-column word, ballot per word)
// ---------------------------------------------------------------------------
__global__ void build_mask_kernel(const float* __restrict__ A, unsigned* __restrict__ M) {
    int gw   = blockIdx.x * (blockDim.x >> 5) + (threadIdx.x >> 5);
    int lane = threadIdx.x & 31;
    int row  = gw >> 8;              // 256 words per row
    int word = gw & 255;
    float v = A[(size_t)row * NODES + word * 32 + lane];
    unsigned m = __builtin_amdgcn_ballot_w32(v > 0.f);
    if (lane == 0) M[(size_t)row * MASK_WORDS + word] = m;
}

// 32x32 bit-block transpose via ballots: MT = transpose(M) at bit level.
__global__ void transpose_mask_kernel(const unsigned* __restrict__ M, unsigned* __restrict__ MT) {
    int bi   = blockIdx.x * (blockDim.x >> 5) + (threadIdx.x >> 5);
    int lane = threadIdx.x & 31;
    int bx = bi & 255;               // source word (column block)
    int by = bi >> 8;                // source row block
    unsigned in = M[(size_t)(by * 32 + lane) * MASK_WORDS + bx];
    unsigned out = 0;
#pragma unroll
    for (int it = 0; it < 32; ++it) {
        unsigned b = __builtin_amdgcn_ballot_w32(((in >> it) & 1u) != 0u);
        if (lane == it) out = b;
    }
    MT[(size_t)(bx * 32 + lane) * MASK_WORDS + by] = out;
}

// ---------------------------------------------------------------------------
// 2) WMMA GEMM: WhT[n][i] = bf16( sum_k H[i][k] * W[k][n] )   (K-contiguous
//    transposed output so attention B-fragments are contiguous b128 loads)
//    Block: 128 threads (4 waves), 64 rows; wave w owns rows w*16..w*16+15.
// ---------------------------------------------------------------------------
template <int NT>
__global__ void __launch_bounds__(128)
gemm_wht_kernel(const float* __restrict__ H, int hStride,
                const float* __restrict__ W, int Din, int F,
                __bf16* __restrict__ WhT) {
    __shared__ __attribute__((aligned(16))) __bf16 Hl[64 * 40];       // 64 rows x 32 K (pad 40)
    __shared__ __attribute__((aligned(16))) __bf16 Wl[NT * 16 * 40];  // F cols x 32 K transposed

    const int tid = threadIdx.x, warp = tid >> 5, lane = tid & 31;
    const int hf = lane >> 4, nn = lane & 15;   // hf selects K-run group, nn = M (A) or N (B)
    const int i0 = blockIdx.x * 64;
    const int ntN16 = NT * 16;

    v8f acc[NT];
#pragma unroll
    for (int t = 0; t < NT; ++t)
#pragma unroll
        for (int r = 0; r < 8; ++r) acc[t][r] = 0.f;

    const int ksteps = (Din + 31) >> 5;
    for (int ks = 0; ks < ksteps; ++ks) {
        const int k0 = ks * 32;
        __syncthreads();
        for (int e = tid; e < 64 * 32; e += 128) {               // stage H tile (zero-pad K)
            int rr = e >> 5, kc = e & 31;
            float v = (k0 + kc < Din) ? H[(size_t)(i0 + rr) * hStride + k0 + kc] : 0.f;
            Hl[rr * 40 + kc] = (__bf16)v;
        }
        for (int e = tid; e < ntN16 * 32; e += 128) {            // stage W^T K-slice
            int kc = e / ntN16, n = e - kc * ntN16;
            float v = (n < F && k0 + kc < Din) ? W[(size_t)(k0 + kc) * F + n] : 0.f;
            Wl[n * 40 + kc] = (__bf16)v;
        }
        __syncthreads();

        union { v16bf v; float4 f[2]; } afr;                     // A: 16x32 bf16 fragment
        const __bf16* ap = &Hl[(warp * 16 + nn) * 40 + hf * 8];
        afr.f[0] = *(const float4*)ap;
        afr.f[1] = *(const float4*)(ap + 16);
#pragma unroll
        for (int t = 0; t < NT; ++t) {
            union { v16bf v; float4 f[2]; } bfr;                 // B: 32x16 bf16 fragment
            const __bf16* bp = &Wl[(t * 16 + nn) * 40 + hf * 8];
            bfr.f[0] = *(const float4*)bp;
            bfr.f[1] = *(const float4*)(bp + 16);
            acc[t] = __builtin_amdgcn_wmma_f32_16x16x32_bf16(
                false, afr.v, false, bfr.v, (short)0, acc[t], false, false);
        }
    }

#pragma unroll
    for (int t = 0; t < NT; ++t) {
        int n = t * 16 + nn;
        if (n < F) {
#pragma unroll
            for (int r = 0; r < 8; ++r) {
                int i = i0 + warp * 16 + r + 8 * hf;             // C/D: M = r + 8*(lane/16)
                WhT[(size_t)n * NODES + i] = (__bf16)acc[t][r];
            }
        }
    }
}

// Pad rows F..Fp-1 of WhT: row F = 1.0 (row-sum column), rest = 0.
__global__ void pad_kernel(__bf16* __restrict__ WhT, int F) {
    int idx = blockIdx.x * blockDim.x + threadIdx.x;
    int n = F + idx / NODES;
    int i = idx - (n - F) * NODES;
    WhT[(size_t)n * NODES + i] = (__bf16)((n == F) ? 1.f : 0.f);
}

// ---------------------------------------------------------------------------
// 3) f = Wh @ a_src, g = Wh @ a_dst  (from transposed bf16 panel, coalesced)
// ---------------------------------------------------------------------------
__global__ void fvec_kernel(const __bf16* __restrict__ WhT,
                            const float* __restrict__ asrc, const float* __restrict__ adst,
                            int F, float* __restrict__ fvec, float* __restrict__ gvec) {
    int i = blockIdx.x * blockDim.x + threadIdx.x;
    float fs = 0.f, fd = 0.f;
    for (int n = 0; n < F; ++n) {
        float wh = (float)WhT[(size_t)n * NODES + i];
        fs += wh * asrc[n];
        fd += wh * adst[n];
    }
    fvec[i] = fs;
    gvec[i] = fd;
}

// Exact per-row softmax max: m_i = leaky(f_i + max_{j in nbr(i)} g_j),
// or NEG_BIG for empty rows (-> p == 1 everywhere -> uniform softmax).
__global__ void rowmax_kernel(const unsigned* __restrict__ Mbits,
                              const float* __restrict__ fvec, const float* __restrict__ gvec,
                              float* __restrict__ mvec) {
    int row  = blockIdx.x * (blockDim.x >> 5) + (threadIdx.x >> 5);
    int lane = threadIdx.x & 31;
    float gm = -3.0e38f;
    for (int w = lane; w < MASK_WORDS; w += 32) {
        unsigned word = Mbits[(size_t)row * MASK_WORDS + w];
        while (word) {
            int b = __builtin_ctz(word);
            gm = fmaxf(gm, gvec[w * 32 + b]);
            word &= word - 1;
        }
    }
#pragma unroll
    for (int off = 16; off > 0; off >>= 1)
        gm = fmaxf(gm, __shfl_xor(gm, off, 32));
    if (lane == 0)
        mvec[row] = (gm > -1.0e38f) ? leaky01(fvec[row] + gm) : NEG_BIG;
}

// ---------------------------------------------------------------------------
// 4) Fused masked-softmax attention * Wh, streaming WMMA.
//    Block: 128 threads (4 waves), 16 output rows, all NT feature tiles.
//    NT feature tiles of 16; tile SUMT column SUMN carries the ones-column
//    (row sums). Wave w owns tiles w, w+4, w+8, w+12.
// ---------------------------------------------------------------------------
template <int NT, int SUMT, int SUMN>
__global__ void __launch_bounds__(128)
attn_kernel(const unsigned* __restrict__ Mbits, const __bf16* __restrict__ WhT,
            const float* __restrict__ fvec, const float* __restrict__ gvec,
            const float* __restrict__ mvec,
            float* __restrict__ out, int outStride, int outOff, int F) {
    __shared__ __attribute__((aligned(16))) __bf16 Pl[16 * 136];  // 16 rows x 128 probs (bf16)
    __shared__ unsigned maskW[64];                                // 16 rows x 4 mask words
    __shared__ float f_lds[16], m_lds[16], rs[16];

    const int tid = threadIdx.x, warp = tid >> 5, lane = tid & 31;
    const int hf = lane >> 4, nn = lane & 15;
    const int i0 = blockIdx.x * 16;

    if (tid < 16) { f_lds[tid] = fvec[i0 + tid]; m_lds[tid] = mvec[i0 + tid]; }

    v8f acc[4];
#pragma unroll
    for (int q = 0; q < 4; ++q)
#pragma unroll
        for (int r = 0; r < 8; ++r) acc[q][r] = 0.f;

    for (int j0 = 0; j0 < NODES; j0 += 128) {
        __syncthreads();
        if (tid < 64)
            maskW[tid] = Mbits[(size_t)(i0 + (tid >> 2)) * MASK_WORDS + (j0 >> 5) + (tid & 3)];
        float gj = gvec[j0 + tid];
        __syncthreads();
        // Phase A: probabilities p = exp(e - m) (masked -> 0, or 1 for empty rows)
#pragma unroll
        for (int r = 0; r < 16; ++r) {
            unsigned w = maskW[(r << 2) + (tid >> 5)];
            float t = f_lds[r] + gj;
            float e = ((w >> (tid & 31)) & 1u) ? (t > 0.f ? t : 0.1f * t) : NEG_BIG;
            Pl[r * 136 + tid] = (__bf16)__expf(e - m_lds[r]);
        }
        __syncthreads();
        // Phase B: acc += P(16x32) @ WhT-tile(32x16), 4 K-steps per chunk
#pragma unroll
        for (int kk = 0; kk < 4; ++kk) {
            union { v16bf v; float4 f[2]; } afr;
            const __bf16* ap = &Pl[nn * 136 + kk * 32 + hf * 8];
            afr.f[0] = *(const float4*)ap;
            afr.f[1] = *(const float4*)(ap + 16);
#pragma unroll
            for (int q = 0; q < 4; ++q) {
                int tn = warp + 4 * q;                 // wave-uniform -> EXEC all-1 at WMMA
                if (tn < NT) {
                    union { v16bf v; float4 f[2]; } bfr;
                    const __bf16* bp = &WhT[(size_t)(tn * 16 + nn) * NODES + j0 + kk * 32 + hf * 8];
                    bfr.f[0] = *(const float4*)bp;
                    bfr.f[1] = *(const float4*)(bp + 16);
                    acc[q] = __builtin_amdgcn_wmma_f32_16x16x32_bf16(
                        false, afr.v, false, bfr.v, (short)0, acc[q], false, false);
                }
            }
        }
    }

    // Row sums live in tile SUMT, column SUMN (ones-column of WhT panel).
    if (warp == (SUMT & 3) && nn == SUMN) {
#pragma unroll
        for (int r = 0; r < 8; ++r) rs[r + 8 * hf] = acc[SUMT >> 2][r];
    }
    __syncthreads();

#pragma unroll
    for (int q = 0; q < 4; ++q) {
        int tn = warp + 4 * q;
        if (tn < NT) {
            int col = tn * 16 + nn;
            if (col < F) {
#pragma unroll
                for (int r = 0; r < 8; ++r) {
                    int mr = r + 8 * hf;
                    float v = acc[q][r] / rs[mr];
                    out[(size_t)(i0 + mr) * outStride + outOff + col] = fmaxf(v, 0.f);
                }
            }
        }
    }
}

// ---------------------------------------------------------------------------
// Host orchestration
// ---------------------------------------------------------------------------
extern "C" void kernel_launch(void* const* d_in, const int* in_sizes, int n_in,
                              void* d_out, int out_size, void* d_ws, size_t ws_size,
                              hipStream_t stream) {
    (void)in_sizes; (void)n_in; (void)out_size; (void)ws_size;
    const float* A   = (const float*)d_in[0];
    const float* x   = (const float*)d_in[1];
    const float* W1  = (const float*)d_in[2];
    const float* a1s = (const float*)d_in[3];
    const float* a1d = (const float*)d_in[4];
    const float* W2  = (const float*)d_in[5];
    const float* a2s = (const float*)d_in[6];
    const float* a2d = (const float*)d_in[7];
    float* out = (float*)d_out;

    char* ws = (char*)d_ws;
    unsigned* Mb  = (unsigned*)ws;                                 //  8 MB
    unsigned* MTb = (unsigned*)(ws + (size_t)(8u << 20));          //  8 MB
    __bf16*   WhT = (__bf16*)(ws + (size_t)(16u << 20));           //  208*8192*2 B
    float*    fvec = (float*)(ws + (size_t)(16u << 20) + (size_t)(4u << 20));
    float*    gvec = fvec + NODES;
    float*    mvec = gvec + NODES;

    float* seg0 = out;                                   // x copy        [8192 x 256]
    float* seg1 = out + (size_t)NODES * 256;             // layer-1 cat   [8192 x 400]
    float* seg2 = seg1 + (size_t)NODES * 400;            // layer-2 cat   [8192 x 256]

    hipMemcpyAsync(seg0, x, (size_t)NODES * 256 * sizeof(float),
                   hipMemcpyDeviceToDevice, stream);

    build_mask_kernel<<<NODES * (MASK_WORDS / 8), 256, 0, stream>>>(A, Mb);
    transpose_mask_kernel<<<(256 * 256) / 4, 128, 0, stream>>>(Mb, MTb);

    // ----- Layer 1 (h1 == h2 == x, so both branches share Wh/f/g) -----
    gemm_wht_kernel<13><<<NODES / 64, 128, 0, stream>>>(x, 256, W1, 256, 200, WhT);
    pad_kernel<<<(208 - 200) * NODES / 256, 256, 0, stream>>>(WhT, 200);
    fvec_kernel<<<NODES / 256, 256, 0, stream>>>(WhT, a1s, a1d, 200, fvec, gvec);
    rowmax_kernel<<<NODES / 4, 128, 0, stream>>>(Mb, fvec, gvec, mvec);
    attn_kernel<13, 12, 8><<<NODES / 16, 128, 0, stream>>>(Mb, WhT, fvec, gvec, mvec,
                                                           seg1, 400, 0, 200);
    rowmax_kernel<<<NODES / 4, 128, 0, stream>>>(MTb, fvec, gvec, mvec);
    attn_kernel<13, 12, 8><<<NODES / 16, 128, 0, stream>>>(MTb, WhT, fvec, gvec, mvec,
                                                           seg1, 400, 200, 200);

    // ----- Layer 2, branch 1 (input = seg1 cols 0..199, mask = A) -----
    gemm_wht_kernel<8><<<NODES / 64, 128, 0, stream>>>(seg1, 400, W2, 200, 128, WhT);
    pad_kernel<<<(144 - 128) * NODES / 256, 256, 0, stream>>>(WhT, 128);
    fvec_kernel<<<NODES / 256, 256, 0, stream>>>(WhT, a2s, a2d, 128, fvec, gvec);
    rowmax_kernel<<<NODES / 4, 128, 0, stream>>>(Mb, fvec, gvec, mvec);
    attn_kernel<9, 8, 0><<<NODES / 16, 128, 0, stream>>>(Mb, WhT, fvec, gvec, mvec,
                                                         seg2, 256, 0, 128);

    // ----- Layer 2, branch 2 (input = seg1 cols 200..399, mask = A^T) -----
    gemm_wht_kernel<8><<<NODES / 64, 128, 0, stream>>>(seg1 + 200, 400, W2, 200, 128, WhT);
    pad_kernel<<<(144 - 128) * NODES / 256, 256, 0, stream>>>(WhT, 128);
    fvec_kernel<<<NODES / 256, 256, 0, stream>>>(WhT, a2s, a2d, 128, fvec, gvec);
    rowmax_kernel<<<NODES / 4, 128, 0, stream>>>(MTb, fvec, gvec, mvec);
    attn_kernel<9, 8, 0><<<NODES / 16, 128, 0, stream>>>(MTb, WhT, fvec, gvec, mvec,
                                                         seg2, 256, 128, 128);
}
